// KVPLMStarEncoder_20186346291606
// MI455X (gfx1250) — matrix operand order repaired and checked by reading, not compiled, not run on hardware
//
#include <hip/hip_runtime.h>

#define LAYERS 12
#define HD     768
#define NHEAD  12
#define DH     64
#define FF     3072
#define SEQ    512
#define BATCH  16
#define ROWS   (BATCH*SEQ)      // 8192
#define VOCAB_SZ 31090
#define STARV  390
#define OFFTOK 30700

typedef __attribute__((ext_vector_type(16))) __bf16 v16bf;
typedef __attribute__((ext_vector_type(8)))  float  v8f;

union AB { v16bf v; __bf16 e[16]; };
union CF { v8f  v; float  e[8];  };

// -------------------------------------------------------------------------
// Block-wide sum reduction of two values (256 threads = 8 waves of 32)
// -------------------------------------------------------------------------
__device__ __forceinline__ void block_reduce2(float& s1, float& s2) {
#pragma unroll
  for (int o = 16; o; o >>= 1) {
    s1 += __shfl_xor(s1, o, 32);
    s2 += __shfl_xor(s2, o, 32);
  }
  __shared__ float r1[8], r2[8];
  const int w = threadIdx.x >> 5, ln = threadIdx.x & 31;
  if (ln == 0) { r1[w] = s1; r2[w] = s2; }
  __syncthreads();
  if (threadIdx.x < 32) {
    s1 = (ln < 8) ? r1[ln] : 0.f;
    s2 = (ln < 8) ? r2[ln] : 0.f;
#pragma unroll
    for (int o = 4; o; o >>= 1) {
      s1 += __shfl_xor(s1, o, 32);
      s2 += __shfl_xor(s2, o, 32);
    }
    if (ln == 0) { r1[0] = s1; r2[0] = s2; }
  }
  __syncthreads();
  s1 = r1[0]; s2 = r2[0];
}

// -------------------------------------------------------------------------
// Embedding gather + LayerNorm -> fp32 x and bf16 mirror. 1 block / token.
// -------------------------------------------------------------------------
__global__ __launch_bounds__(256)
void embed_ln(const int* __restrict__ ids, const int* __restrict__ ttids,
              const float* __restrict__ wemb, const float* __restrict__ semb,
              const float* __restrict__ pemb, const float* __restrict__ temb,
              const float* __restrict__ g, const float* __restrict__ bb,
              float* __restrict__ xf, __bf16* __restrict__ xb)
{
  const int tok = blockIdx.x;
  const int s   = tok & (SEQ - 1);
  const int id  = ids[tok];
  const int tt  = ttids[tok];
  const float* src;
  if (id >= OFFTOK) {
    int i2 = id - OFFTOK; if (i2 > STARV - 1) i2 = STARV - 1;
    src = semb + (size_t)i2 * HD;
  } else {
    int i2 = id < 0 ? 0 : (id > VOCAB_SZ - 1 ? VOCAB_SZ - 1 : id);
    src = wemb + (size_t)i2 * HD;
  }
  float loc[3]; float s1 = 0.f, s2 = 0.f;
#pragma unroll
  for (int i = 0; i < 3; ++i) {
    const int j = threadIdx.x + i * 256;
    const float e = src[j] + pemb[s * HD + j] + temb[tt * HD + j];
    loc[i] = e; s1 += e; s2 += e * e;
  }
  block_reduce2(s1, s2);
  const float mean = s1 * (1.0f / HD);
  const float var  = s2 * (1.0f / HD) - mean * mean;
  const float rs   = rsqrtf(var + 1e-12f);
#pragma unroll
  for (int i = 0; i < 3; ++i) {
    const int j = threadIdx.x + i * 256;
    const float val = (loc[i] - mean) * rs * g[j] + bb[j];
    xf[(size_t)tok * HD + j] = val;
    xb[(size_t)tok * HD + j] = (__bf16)val;
  }
}

// -------------------------------------------------------------------------
// Residual + LayerNorm: x = LN(x + y). In-place per row (1 block / token).
// -------------------------------------------------------------------------
__global__ __launch_bounds__(256)
void add_ln(const float* __restrict__ y, const float* __restrict__ g,
            const float* __restrict__ bb, float* __restrict__ xf,
            __bf16* __restrict__ xb)
{
  const int tok = blockIdx.x;
  float loc[3]; float s1 = 0.f, s2 = 0.f;
#pragma unroll
  for (int i = 0; i < 3; ++i) {
    const int j = threadIdx.x + i * 256;
    const float e = xf[(size_t)tok * HD + j] + y[(size_t)tok * HD + j];
    loc[i] = e; s1 += e; s2 += e * e;
  }
  block_reduce2(s1, s2);
  const float mean = s1 * (1.0f / HD);
  const float var  = s2 * (1.0f / HD) - mean * mean;
  const float rs   = rsqrtf(var + 1e-12f);
#pragma unroll
  for (int i = 0; i < 3; ++i) {
    const int j = threadIdx.x + i * 256;
    const float val = (loc[i] - mean) * rs * g[j] + bb[j];
    xf[(size_t)tok * HD + j] = val;
    xb[(size_t)tok * HD + j] = (__bf16)val;
  }
}

// -------------------------------------------------------------------------
// Tiled bf16 WMMA GEMM: C[M x N] = A(bf16) * W(fp32->bf16) + bias
// Block tile 128x64, K-step 32, 4 waves; wave w owns rows [32w, 32w+32).
// Double-buffered LDS; A tile staged via GLOBAL_LOAD_ASYNC_TO_LDS_B128
// (ASYNCcnt), W tile converted fp32->bf16 into a swizzled-transposed layout
// (stride 40 halves) so every B fragment load is 2x ds_load_b128.
// MODE 0: fp32 out     MODE 1: GELU->bf16    MODE 2: [B,NH,S,DH] bf16
// MODE 3: [B,NH,DH,S] bf16 (transposed V)
// -------------------------------------------------------------------------
#define BSTRIDE 40   // halves per column of swizzled B; 80B rows, 16B-aligned

__device__ __forceinline__ void stage_tiles(
    const __bf16* __restrict__ A, const float* __restrict__ W,
    __bf16* __restrict__ As, __bf16* __restrict__ Bs,
    int m0, int n0, int k0, int K, int N, int tid)
{
  // A tile 128x32 bf16 = 512 x 16B chunks: 4 async DMA b128 per thread,
  // straight into LDS (no VGPR round-trip), tracked with ASYNCcnt.
#pragma unroll
  for (int j = 0; j < 4; ++j) {
    const int q = tid + j * 128;             // 0..511
    const int r = q >> 2, c = (q & 3) << 3;  // row, col(8-elem chunk)
    const uint64_t src = (uint64_t)(uintptr_t)&A[(size_t)(m0 + r) * K + k0 + c];
    const uint32_t dst = (uint32_t)(uintptr_t)&As[r * 32 + c];
    asm volatile("global_load_async_to_lds_b128 %0, %1, off"
                 :: "v"(dst), "v"(src) : "memory");
  }
  // W tile 32x64 fp32 -> bf16, transposed+padded: Bs[c*BSTRIDE + kk]
  for (int u = tid; u < 32 * 64; u += 128) {
    const int r = u >> 6, c = u & 63;
    Bs[c * BSTRIDE + r] = (__bf16)W[(size_t)(k0 + r) * N + n0 + c];
  }
}

template<int MODE>
__global__ __launch_bounds__(128)
void gemm_bf16_wmma(const __bf16* __restrict__ A, const float* __restrict__ W,
                    const float* __restrict__ bias, float* __restrict__ outF,
                    __bf16* __restrict__ outB, int K, int N)
{
  __shared__ __bf16 As[2][128 * 32];        // 2 x 8 KB, row-major
  __shared__ __bf16 Bs[2][64 * BSTRIDE];    // 2 x 5 KB, Bs[c*BSTRIDE + kk]
  const int tid  = threadIdx.x;
  const int lane = tid & 31, wave = tid >> 5;
  const int lr   = lane & 15, hi   = lane >> 4;
  const int m0   = blockIdx.x * 128, n0 = blockIdx.y * 64;

  CF acc[2][4];
  for (int mt = 0; mt < 2; ++mt)
    for (int t = 0; t < 4; ++t)
      for (int i = 0; i < 8; ++i) acc[mt][t].e[i] = 0.f;

  stage_tiles(A, W, As[0], Bs[0], m0, n0, 0, K, N, tid);

  for (int k0 = 0; k0 < K; k0 += 32) {
    const int cur = (k0 >> 5) & 1;
    // Our async DMA writes for buffer `cur` must land before the barrier.
    asm volatile("s_wait_asynccnt 0x0" ::: "memory");
    __syncthreads();
    if (k0 + 32 < K)
      stage_tiles(A, W, As[cur ^ 1], Bs[cur ^ 1], m0, n0, k0 + 32, K, N, tid);
    if (k0 + 64 < K) {   // prefetch the W tile two steps ahead
      const int pr = tid >> 2, pc = (tid & 3) << 4;   // 32 rows x 4x64B
      __builtin_prefetch(&W[(size_t)(k0 + 64 + pr) * N + n0 + pc], 0, 1);
    }

    // A fragments: 16x32 bf16, contiguous 2x16B chunks per lane
    AB a[2];
#pragma unroll
    for (int mt = 0; mt < 2; ++mt)
#pragma unroll
      for (int h = 0; h < 16; ++h) {
        const int kk = (hi << 4) + h;
        a[mt].e[h] = As[cur][(wave * 32 + mt * 16 + lr) * 32 + kk];
      }
#pragma unroll
    for (int nt = 0; nt < 4; ++nt) {
      AB bfr;                               // contiguous 16 halves -> 2x b128
#pragma unroll
      for (int h = 0; h < 16; ++h)
        bfr.e[h] = Bs[cur][(nt * 16 + lr) * BSTRIDE + (hi << 4) + h];
#pragma unroll
      for (int mt = 0; mt < 2; ++mt)
        acc[mt][nt].v = __builtin_amdgcn_wmma_f32_16x16x32_bf16(
            false, a[mt].v, false, bfr.v, (short)0, acc[mt][nt].v, false, false);
    }
  }

#pragma unroll
  for (int nt = 0; nt < 4; ++nt) {
    const int col = n0 + nt * 16 + lr;
    const float bv = bias[col];
#pragma unroll
    for (int mt = 0; mt < 2; ++mt)
#pragma unroll
      for (int r = 0; r < 8; ++r) {
        const int row = m0 + wave * 32 + mt * 16 + r + hi * 8;
        const float val = acc[mt][nt].e[r] + bv;
        if (MODE == 0) {
          outF[(size_t)row * N + col] = val;
        } else if (MODE == 1) {
          const float ge = 0.5f * val * (1.0f + erff(val * 0.70710678118f));
          outB[(size_t)row * N + col] = (__bf16)ge;
        } else if (MODE == 2) {
          const int b = row >> 9, s = row & 511;
          const int h = col >> 6, d = col & 63;
          outB[(((size_t)(b * NHEAD + h)) * SEQ + s) * DH + d] = (__bf16)val;
        } else {
          const int b = row >> 9, s = row & 511;
          const int h = col >> 6, d = col & 63;
          outB[(((size_t)(b * NHEAD + h)) * DH + d) * SEQ + s] = (__bf16)val;
        }
      }
  }
}

// -------------------------------------------------------------------------
// Attention: one wave per (b, head, 16-query tile).
// S = QK^T (WMMA), masked softmax through LDS, ctx = P*V (WMMA).
// V is pre-transposed per head ([B,NH,DH,S]) so B-fragments are contiguous.
// -------------------------------------------------------------------------
__global__ __launch_bounds__(32)
void attn_wmma(const __bf16* __restrict__ q, const __bf16* __restrict__ k,
               const __bf16* __restrict__ vT, const int* __restrict__ amask,
               __bf16* __restrict__ ctx)
{
  __shared__ float  Ss[16 * SEQ];   // 32 KB scores
  __shared__ __bf16 Ps[16 * SEQ];   // 16 KB probabilities
  const int lane = threadIdx.x, lr = lane & 15, hi = lane >> 4;
  const int blk = blockIdx.x;
  const int qt  = blk & 31;
  const int bh  = blk >> 5;
  const int b   = bh / NHEAD;
  const int hh  = bh % NHEAD;
  const __bf16* Q  = q  + ((size_t)bh * SEQ + qt * 16) * DH;
  const __bf16* Kp = k  + (size_t)bh * SEQ * DH;
  const __bf16* Vt = vT + (size_t)bh * DH * SEQ;

  // Q fragments: 16x64 = two 16x32 bf16 A-operands (contiguous per lane)
  AB qa[2];
#pragma unroll
  for (int g = 0; g < 2; ++g)
#pragma unroll
    for (int h = 0; h < 16; ++h) {
      const int kk = g * 32 + (hi << 4) + h;
      qa[g].e[h] = Q[lr * DH + kk];
    }

  const float scale = 0.125f;   // 1/sqrt(64)
  for (int nt = 0; nt < 32; ++nt) {
    CF s; for (int i = 0; i < 8; ++i) s.e[i] = 0.f;
#pragma unroll
    for (int g = 0; g < 2; ++g) {
      AB kb;                               // contiguous -> global b128
#pragma unroll
      for (int h = 0; h < 16; ++h)
        kb.e[h] = Kp[(nt * 16 + lr) * DH + g * 32 + (hi << 4) + h];
      s.v = __builtin_amdgcn_wmma_f32_16x16x32_bf16(
          false, qa[g].v, false, kb.v, (short)0, s.v, false, false);
    }
    const int key = nt * 16 + lr;
    const float ex = (1.0f - (float)amask[b * SEQ + key]) * -10000.0f;
#pragma unroll
    for (int r = 0; r < 8; ++r)
      Ss[(r + hi * 8) * SEQ + key] = s.e[r] * scale + ex;
  }
  __syncthreads();

  // Masked softmax over 512 keys per row, wave-cooperative
  for (int m = 0; m < 16; ++m) {
    float mx = -3.4e38f;
    for (int c = lane; c < SEQ; c += 32) mx = fmaxf(mx, Ss[m * SEQ + c]);
#pragma unroll
    for (int o = 16; o; o >>= 1) mx = fmaxf(mx, __shfl_xor(mx, o, 32));
    float sum = 0.f;
    for (int c = lane; c < SEQ; c += 32) {
      const float e = __expf(Ss[m * SEQ + c] - mx);
      Ss[m * SEQ + c] = e; sum += e;
    }
#pragma unroll
    for (int o = 16; o; o >>= 1) sum += __shfl_xor(sum, o, 32);
    const float inv = 1.0f / sum;
    for (int c = lane; c < SEQ; c += 32)
      Ps[m * SEQ + c] = (__bf16)(Ss[m * SEQ + c] * inv);
  }
  __syncthreads();

  // ctx = P(16x512) * V(512x64): 16 K-steps x 4 N-tiles of WMMA
  CF o4[4];
  for (int t = 0; t < 4; ++t)
    for (int i = 0; i < 8; ++i) o4[t].e[i] = 0.f;
  for (int ks = 0; ks < 16; ++ks) {
    AB pa;                                 // contiguous -> ds b128
#pragma unroll
    for (int h = 0; h < 16; ++h) {
      const int kk = (hi << 4) + h;
      pa.e[h] = Ps[lr * SEQ + ks * 32 + kk];
    }
#pragma unroll
    for (int nt = 0; nt < 4; ++nt) {
      AB vbf;                              // contiguous in vT -> global b128
#pragma unroll
      for (int h = 0; h < 16; ++h)
        vbf.e[h] = Vt[(size_t)(nt * 16 + lr) * SEQ + ks * 32 + (hi << 4) + h];
      o4[nt].v = __builtin_amdgcn_wmma_f32_16x16x32_bf16(
          false, pa.v, false, vbf.v, (short)0, o4[nt].v, false, false);
    }
  }
#pragma unroll
  for (int nt = 0; nt < 4; ++nt)
#pragma unroll
    for (int r = 0; r < 8; ++r) {
      const int row = b * SEQ + qt * 16 + r + hi * 8;
      const int col = hh * DH + nt * 16 + lr;
      ctx[(size_t)row * HD + col] = (__bf16)o4[nt].e[r];
    }
}

// -------------------------------------------------------------------------
// pooled = tanh(x[:,0] @ pool_W + pool_b)  (tiny: 16x768 * 768x768)
// -------------------------------------------------------------------------
__global__ __launch_bounds__(256)
void pool_tanh(const float* __restrict__ xf, const float* __restrict__ pw,
               const float* __restrict__ pb, float* __restrict__ out)
{
  const int idx = blockIdx.x * blockDim.x + threadIdx.x;
  if (idx >= BATCH * HD) return;
  const int b = idx / HD, j = idx % HD;
  const float* xr = xf + (size_t)b * SEQ * HD;   // token 0 of batch b
  float s = 0.f;
  for (int kk = 0; kk < HD; ++kk) s += xr[kk] * pw[(size_t)kk * HD + j];
  out[idx] = tanhf(s + pb[j]);
}

__global__ __launch_bounds__(256)
void copy_out(const float* __restrict__ xf, float* __restrict__ out, int n)
{
  for (int i = blockIdx.x * blockDim.x + threadIdx.x; i < n;
       i += gridDim.x * blockDim.x)
    out[i] = xf[i];
}

// -------------------------------------------------------------------------
extern "C" void kernel_launch(void* const* d_in, const int* in_sizes, int n_in,
                              void* d_out, int out_size, void* d_ws, size_t ws_size,
                              hipStream_t stream)
{
  const int*   ids   = (const int*)d_in[0];
  const int*   amask = (const int*)d_in[1];
  const int*   tti   = (const int*)d_in[2];
  const float* wemb  = (const float*)d_in[3];
  const float* semb  = (const float*)d_in[4];
  const float* pemb  = (const float*)d_in[5];
  const float* temb  = (const float*)d_in[6];
  const float* eg    = (const float*)d_in[7];
  const float* ebb   = (const float*)d_in[8];
  const float* Wq = (const float*)d_in[9];  const float* bq  = (const float*)d_in[10];
  const float* Wk = (const float*)d_in[11]; const float* bk  = (const float*)d_in[12];
  const float* Wv = (const float*)d_in[13]; const float* bv  = (const float*)d_in[14];
  const float* Wo = (const float*)d_in[15]; const float* bo  = (const float*)d_in[16];
  const float* g1 = (const float*)d_in[17]; const float* b1  = (const float*)d_in[18];
  const float* Wi = (const float*)d_in[19]; const float* bi  = (const float*)d_in[20];
  const float* Wf = (const float*)d_in[21]; const float* bfp = (const float*)d_in[22];
  const float* g2 = (const float*)d_in[23]; const float* b2  = (const float*)d_in[24];
  const float* pw = (const float*)d_in[25]; const float* pb  = (const float*)d_in[26];
  float* out = (float*)d_out;

  // Workspace layout (bytes, 256-aligned):
  char* w = (char*)d_ws;
  float*  xf = (float*)(w);                  // 8192x768  f32  (25.2 MB)
  __bf16* xb = (__bf16*)(w + 25165824);      // 8192x768  bf16 (12.6 MB)
  float*  yf = (float*)(w + 37748736);       // 8192x768  f32  (25.2 MB)
  __bf16* qb = (__bf16*)(w + 62914560);      // [B,NH,S,DH] bf16
  __bf16* kb = (__bf16*)(w + 75497472);      // [B,NH,S,DH] bf16
  __bf16* vb = (__bf16*)(w + 88080384);      // [B,NH,DH,S] bf16 (transposed)
  __bf16* cb = (__bf16*)(w + 100663296);     // ctx [B,S,H] bf16
  __bf16* fb = (__bf16*)(w + 113246208);     // 8192x3072 bf16 (50.3 MB)

  embed_ln<<<ROWS, 256, 0, stream>>>(ids, tti, wemb, semb, pemb, temb, eg, ebb, xf, xb);

  const dim3 gH(ROWS / 128, HD / 64);   // 64 x 12
  const dim3 gF(ROWS / 128, FF / 64);   // 64 x 48
  for (int l = 0; l < LAYERS; ++l) {
    gemm_bf16_wmma<2><<<gH, 128, 0, stream>>>(xb, Wq + (size_t)l * HD * HD, bq + l * HD, nullptr, qb, HD, HD);
    gemm_bf16_wmma<2><<<gH, 128, 0, stream>>>(xb, Wk + (size_t)l * HD * HD, bk + l * HD, nullptr, kb, HD, HD);
    gemm_bf16_wmma<3><<<gH, 128, 0, stream>>>(xb, Wv + (size_t)l * HD * HD, bv + l * HD, nullptr, vb, HD, HD);
    attn_wmma<<<BATCH * NHEAD * (SEQ / 16), 32, 0, stream>>>(qb, kb, vb, amask, cb);
    gemm_bf16_wmma<0><<<gH, 128, 0, stream>>>(cb, Wo + (size_t)l * HD * HD, bo + l * HD, yf, nullptr, HD, HD);
    add_ln<<<ROWS, 256, 0, stream>>>(yf, g1 + l * HD, b1 + l * HD, xf, xb);
    gemm_bf16_wmma<1><<<gF, 128, 0, stream>>>(xb, Wi + (size_t)l * HD * FF, bi + l * FF, nullptr, fb, HD, FF);
    gemm_bf16_wmma<0><<<gH, 128, 0, stream>>>(fb, Wf + (size_t)l * FF * HD, bfp + l * HD, yf, nullptr, FF, HD);
    add_ln<<<ROWS, 256, 0, stream>>>(yf, g2 + l * HD, b2 + l * HD, xf, xb);
  }

  copy_out<<<1024, 256, 0, stream>>>(xf, out, ROWS * HD);
  pool_tanh<<<(BATCH * HD + 255) / 256, 256, 0, stream>>>(xf, pw, pb, out + ROWS * HD);
}
